// SelfAttention_59253368815867
// MI455X (gfx1250) — matrix operand order repaired and checked by reading, not compiled, not run on hardware
//
#include <hip/hip_runtime.h>
#include <hip/hip_bf16.h>

// ---------------------------------------------------------------------------
// Self-attention (B=4, H=W=64, C=256, Cqk=32) for gfx1250 (MI455X).
// bf16 WMMA flash-attention with block-level async LDS staging of K/V tiles,
// key-split partials + combine. f32 accumulate everywhere.
// ---------------------------------------------------------------------------

typedef __attribute__((ext_vector_type(16))) __bf16 v16bf;
typedef __attribute__((ext_vector_type(8)))  __bf16 v8bf;
typedef __attribute__((ext_vector_type(8)))  float  v8f;
typedef __attribute__((ext_vector_type(4)))  int    v4i;

typedef __attribute__((address_space(1))) v4i* as1_v4i;
typedef __attribute__((address_space(3))) v4i* as3_v4i;

#define NTOK   4096
#define NBATCH 4
#define CDIM   256
#define CQK    32
#define ROWS   (NBATCH * NTOK)    // 16384
#define JCOLS  (CQK + CQK + CDIM) // 320
#define QBLK   128                // queries per block (8 waves x 16)
#define KTILE  32                 // keys per inner step
#define KSMAX  4                  // max key-split

// ---- async global->LDS staging (CDNA5 path), with portable fallback -------
#if defined(__has_builtin)
# if __has_builtin(__builtin_amdgcn_global_load_async_to_lds_b128)
#  define HAVE_ASYNC_LDS 1
# endif
#endif

#ifdef HAVE_ASYNC_LDS
// signature (probe-confirmed by diagnostic): (global v4i*, local v4i*, imm, imm)
#define COPY16(gp, lp)                                                        \
    __builtin_amdgcn_global_load_async_to_lds_b128(                           \
        (as1_v4i)(v4i*)(gp), (as3_v4i)(v4i*)(lp), 0, 0)
#else
#define COPY16(gp, lp) do { *(v8bf*)(lp) = *(const v8bf*)(gp); } while (0)
#endif

static __device__ __forceinline__ void wait_async_n5() {
#ifdef HAVE_ASYNC_LDS
# if __has_builtin(__builtin_amdgcn_s_wait_asynccnt)
    __builtin_amdgcn_s_wait_asynccnt(5);
# else
    asm volatile("s_wait_asynccnt 5" ::: "memory");
# endif
#endif
}
static __device__ __forceinline__ void wait_async_n0() {
#ifdef HAVE_ASYNC_LDS
# if __has_builtin(__builtin_amdgcn_s_wait_asynccnt)
    __builtin_amdgcn_s_wait_asynccnt(0);
# else
    asm volatile("s_wait_asynccnt 0" ::: "memory");
# endif
#endif
}

static __device__ __forceinline__ v8bf ld8(const __bf16* p) {
    return *(const v8bf*)p;
}
static __device__ __forceinline__ v16bf cat16(v8bf lo, v8bf hi) {
    return __builtin_shufflevector(lo, hi, 0,1,2,3,4,5,6,7,8,9,10,11,12,13,14,15);
}
static __device__ __forceinline__ v8f wmma_bf16(v16bf a, v16bf b, v8f c) {
    return __builtin_amdgcn_wmma_f32_16x16x32_bf16(
        false, a, false, b, (short)0, c, false, false);
}
static __device__ __forceinline__ float rmax16(float v) {
    #pragma unroll
    for (int m = 8; m >= 1; m >>= 1) v = fmaxf(v, __shfl_xor(v, m, 16));
    return v;
}
static __device__ __forceinline__ float rsum16(float v) {
    #pragma unroll
    for (int m = 8; m >= 1; m >>= 1) v += __shfl_xor(v, m, 16);
    return v;
}

// --------------------------- prep kernels ----------------------------------

__global__ __launch_bounds__(256) void prep_wb_kernel(
    const float* __restrict__ Wf, const float* __restrict__ bfv,
    const float* __restrict__ Wg, const float* __restrict__ bgv,
    const float* __restrict__ Wh, const float* __restrict__ bhv,
    __bf16* __restrict__ Wt, float* __restrict__ bias)
{
    int i = blockIdx.x * 256 + threadIdx.x;
    if (i < JCOLS * CDIM) {
        int j = i / CDIM, c = i % CDIM;
        float w = (j < CQK)     ? Wf[c * CQK + j]
                : (j < 2 * CQK) ? Wg[c * CQK + (j - CQK)]
                                : Wh[c * CDIM + (j - 2 * CQK)];
        Wt[(size_t)j * CDIM + c] = (__bf16)w;
    }
    if (i < JCOLS) {
        bias[i] = (i < CQK) ? bfv[i] : (i < 2 * CQK) ? bgv[i - CQK] : bhv[i - 2 * CQK];
    }
}

__global__ __launch_bounds__(256) void prep_x_kernel(
    const float* __restrict__ x, __bf16* __restrict__ xb)
{
    size_t i = (size_t)blockIdx.x * 256 + threadIdx.x;
    if (i < (size_t)ROWS * CDIM) xb[i] = (__bf16)x[i];
}

// --------------------------- projection GEMM -------------------------------
// f:[B*N][32] g:[B*N][32] (row-major bf16), vt:[B][256][N] (values transposed)
__global__ __launch_bounds__(256) void proj_kernel(
    const __bf16* __restrict__ xb, const __bf16* __restrict__ Wt,
    const float* __restrict__ bias,
    __bf16* __restrict__ fbuf, __bf16* __restrict__ gbuf, __bf16* __restrict__ vt)
{
    const int lane = threadIdx.x & 31;
    const int wv   = threadIdx.x >> 5;
    const int wid  = blockIdx.x * 8 + wv;
    const int rt   = wid / (JCOLS / 16);
    const int ct   = wid % (JCOLS / 16);
    const int half = lane >> 4, l16 = lane & 15;

    const int row = rt * 16 + l16;
    const int col = ct * 16 + l16;

    const __bf16* arow = xb + (size_t)row * CDIM;
    const __bf16* brow = Wt + (size_t)col * CDIM;

    v8f acc = {};
    #pragma unroll
    for (int kt = 0; kt < CDIM / 32; ++kt) {
        const int kb = kt * 32;
        v16bf a = cat16(ld8(arow + kb + half * 16),
                        ld8(arow + kb + half * 16 + 8));
        v16bf b = cat16(ld8(brow + kb + half * 16),
                        ld8(brow + kb + half * 16 + 8));
        acc = wmma_bf16(a, b, acc);
    }

    const float bb = bias[col];
    #pragma unroll
    for (int r = 0; r < 8; ++r) {
        float v = acc[r] + bb;
        int orow = rt * 16 + r + half * 8;
        if (col < CQK) {
            fbuf[(size_t)orow * CQK + col] = (__bf16)v;
        } else if (col < 2 * CQK) {
            gbuf[(size_t)orow * CQK + (col - CQK)] = (__bf16)v;
        } else {
            int b = orow >> 12, n = orow & (NTOK - 1), c = col - 2 * CQK;
            vt[((size_t)b * CDIM + c) * NTOK + n] = (__bf16)v;
        }
    }
}

// --------------------------- flash attention (partials) --------------------
// Block: 8 waves, 128 queries, one key-split range of NTOK/ksplit keys.
// K/V tiles staged once per block into double-buffered LDS (async path).
__global__ __launch_bounds__(256) void attn_kernel(
    const __bf16* __restrict__ fbuf, const __bf16* __restrict__ gbuf,
    const __bf16* __restrict__ vt,
    float* __restrict__ Opart, float* __restrict__ mpart, float* __restrict__ lpart,
    int ksplit)
{
    __shared__ __bf16 sv[2][CDIM * KTILE];   // 2 x 16 KB  value tiles
    __shared__ __bf16 sk[2][KTILE * KTILE];  // 2 x  2 KB  key tiles
    __shared__ __bf16 pstage[8][16 * KTILE]; // 8 KB       per-wave P re-layout

    const int tid  = threadIdx.x;
    const int lane = tid & 31, wv = tid >> 5;
    const int half = lane >> 4, l16 = lane & 15;

    const int ks    = blockIdx.x % ksplit;
    const int bq    = blockIdx.x / ksplit;
    const int b     = bq >> 5;                 // 32 query-blocks per batch
    const int qblk  = bq & 31;
    const int qbase = qblk * QBLK + wv * 16;
    const int kspan = NTOK / ksplit;
    const int k0    = ks * kspan;
    const int T     = kspan / KTILE;

    const __bf16* fb = fbuf + (size_t)b * NTOK * CQK;
    const __bf16* vb = vt   + (size_t)b * CDIM * NTOK;
    const __bf16* gq = gbuf + ((size_t)b * NTOK + qbase) * CQK;
    __bf16* ps = pstage[wv];

    // cooperative stage of one K/V tile: every thread issues exactly 5 copies
    auto stage = [&](int buf, int kb) {
        #pragma unroll
        for (int j = 0; j < 4; ++j) {
            int v = tid + 256 * j;            // 1024 16B chunks of V tile
            int c = v >> 2, part = v & 3;
            COPY16(vb + (size_t)c * NTOK + kb + part * 8,
                   &sv[buf][c * KTILE + part * 8]);
        }
        int kk  = tid & 127;                  // 128 16B chunks of K tile (dup x2)
        int key = kk >> 2, kp = kk & 3;
        COPY16(fb + (size_t)(kb + key) * CQK + kp * 8,
               &sk[buf][key * KTILE + kp * 8]);
    };

    // Q fragment (A layout 16x32): row = l16, K 16-run selected by lane half
    const v16bf Qf = cat16(ld8(gq + l16 * CQK + half * 16),
                           ld8(gq + l16 * CQK + half * 16 + 8));

    float m[8], l[8];
    v8f acc[16];
    #pragma unroll
    for (int r = 0; r < 8; ++r) { m[r] = -3.0e38f; l[r] = 0.0f; }
    #pragma unroll
    for (int ctu = 0; ctu < 16; ++ctu) acc[ctu] = (v8f){};

    stage(0, k0);
    for (int kt = 0; kt < T; ++kt) {
        const int p = kt & 1;
        if (kt + 1 < T) { stage(1 - p, k0 + (kt + 1) * KTILE); wait_async_n5(); }
        else            { wait_async_n0(); }
        __syncthreads();                      // tile kt resident for all waves

        const __bf16* skp = sk[p];
        const __bf16* svp = sv[p];

        // ---- S = Q * K^T for 32 keys ----
        v16bf B0 = cat16(ld8(skp + l16 * KTILE + half * 16),
                         ld8(skp + l16 * KTILE + half * 16 + 8));
        v16bf B1 = cat16(ld8(skp + (16 + l16) * KTILE + half * 16),
                         ld8(skp + (16 + l16) * KTILE + half * 16 + 8));
        v8f z = {};
        v8f s0 = wmma_bf16(Qf, B0, z);
        v8f s1 = wmma_bf16(Qf, B1, z);

        // ---- online softmax (row = r + 8*half, keys striped over 16 lanes) --
        float scale[8];
        #pragma unroll
        for (int r = 0; r < 8; ++r) {
            float t  = rmax16(fmaxf(s0[r], s1[r]));
            float mn = fmaxf(m[r], t);
            float sc = __expf(m[r] - mn);
            float p0 = __expf(s0[r] - mn);
            float p1 = __expf(s1[r] - mn);
            l[r] = l[r] * sc + rsum16(p0 + p1);
            m[r] = mn;
            scale[r] = sc;
            int q = r + half * 8;
            ps[q * KTILE + l16]      = (__bf16)p0;
            ps[q * KTILE + 16 + l16] = (__bf16)p1;
        }
        #pragma unroll
        for (int ctu = 0; ctu < 16; ++ctu)
            #pragma unroll
            for (int r = 0; r < 8; ++r) acc[ctu][r] *= scale[r];

        asm volatile("s_wait_dscnt 0" ::: "memory"); // P stage visible (same wave)

        v16bf Pf = cat16(ld8(ps + l16 * KTILE + half * 16),
                         ld8(ps + l16 * KTILE + half * 16 + 8));

        // ---- O += P * V over 256 channels (16 WMMAs from LDS) ----
        #pragma unroll
        for (int ctu = 0; ctu < 16; ++ctu) {
            const __bf16* vp = svp + (ctu * 16 + l16) * KTILE + half * 16;
            v16bf Bv = cat16(ld8(vp), ld8(vp + 8));
            acc[ctu] = wmma_bf16(Pf, Bv, acc[ctu]);
        }
        __syncthreads();                      // all reads of buf p done
    }

    // ---- write unnormalized partials ----
    #pragma unroll
    for (int r = 0; r < 8; ++r) {
        int q = qbase + r + half * 8;
        size_t rowg = (size_t)b * NTOK + q;
        if (l16 == 0) {
            mpart[(size_t)ks * ROWS + rowg] = m[r];
            lpart[(size_t)ks * ROWS + rowg] = l[r];
        }
        #pragma unroll
        for (int ctu = 0; ctu < 16; ++ctu) {
            Opart[((size_t)ks * ROWS + rowg) * CDIM + ctu * 16 + l16] = acc[ctu][r];
        }
    }
}

// --------------------------- split combine + epilogue ----------------------
__global__ __launch_bounds__(256) void combine_kernel(
    const float* __restrict__ Opart, const float* __restrict__ mpart,
    const float* __restrict__ lpart, const float* __restrict__ x,
    const float* __restrict__ gamma_p, float* __restrict__ y, int ksplit)
{
    size_t i = (size_t)blockIdx.x * 256 + threadIdx.x;
    if (i >= (size_t)ROWS * CDIM) return;
    size_t row = i / CDIM;

    float M = -3.0e38f;
    for (int s = 0; s < ksplit; ++s) M = fmaxf(M, mpart[(size_t)s * ROWS + row]);
    float L = 0.0f, O = 0.0f;
    for (int s = 0; s < ksplit; ++s) {
        float w = __expf(mpart[(size_t)s * ROWS + row] - M);
        L += lpart[(size_t)s * ROWS + row] * w;
        O += Opart[(size_t)s * ROWS * CDIM + i] * w;
    }
    y[i] = gamma_p[0] * (O / L) + x[i];
}

// --------------------------- host launcher ---------------------------------

extern "C" void kernel_launch(void* const* d_in, const int* in_sizes, int n_in,
                              void* d_out, int out_size, void* d_ws, size_t ws_size,
                              hipStream_t stream)
{
    const float* x  = (const float*)d_in[0];
    const float* Wf = (const float*)d_in[1];
    const float* bf = (const float*)d_in[2];
    const float* Wg = (const float*)d_in[3];
    const float* bg = (const float*)d_in[4];
    const float* Wh = (const float*)d_in[5];
    const float* bh = (const float*)d_in[6];
    const float* gm = (const float*)d_in[7];
    float* y = (float*)d_out;

    // workspace layout (256B aligned)
    auto al = [](size_t v) { return (v + 255) & ~(size_t)255; };
    char* base = (char*)d_ws;
    const size_t off_xb   = 0;                                          // 8 MB
    const size_t off_wt   = al(off_xb + (size_t)ROWS * CDIM * 2);
    const size_t off_bias = al(off_wt + (size_t)JCOLS * CDIM * 2);
    const size_t off_f    = al(off_bias + (size_t)JCOLS * 4);           // 1 MB
    const size_t off_g    = al(off_f + (size_t)ROWS * CQK * 2);         // 1 MB
    const size_t off_vt   = al(off_g + (size_t)ROWS * CQK * 2);         // 8 MB
    const size_t off_m    = al(off_vt + (size_t)NBATCH * CDIM * NTOK * 2);
    const size_t off_l    = al(off_m + (size_t)KSMAX * ROWS * 4);
    const size_t off_O    = al(off_l + (size_t)KSMAX * ROWS * 4);
    const size_t opbytes  = (size_t)ROWS * CDIM * 4;                    // 16.8 MB / split

    int ksplit = 1;
    if (ws_size >= off_O + 4 * opbytes)      ksplit = 4;
    else if (ws_size >= off_O + 2 * opbytes) ksplit = 2;

    __bf16* xb    = (__bf16*)(base + off_xb);
    __bf16* Wt    = (__bf16*)(base + off_wt);
    float*  bias  = (float*)(base + off_bias);
    __bf16* fbuf  = (__bf16*)(base + off_f);
    __bf16* gbuf  = (__bf16*)(base + off_g);
    __bf16* vt    = (__bf16*)(base + off_vt);
    float*  mpart = (float*)(base + off_m);
    float*  lpart = (float*)(base + off_l);
    float*  Opart = (float*)(base + off_O);

    // 1) prep
    prep_wb_kernel<<<(JCOLS * CDIM + 255) / 256, 256, 0, stream>>>(
        Wf, bf, Wg, bg, Wh, bh, Wt, bias);
    prep_x_kernel<<<(int)(((size_t)ROWS * CDIM + 255) / 256), 256, 0, stream>>>(x, xb);

    // 2) projections: 1024 row-tiles x 20 col-tiles, 8 waves/block
    proj_kernel<<<(1024 * (JCOLS / 16)) / 8, 256, 0, stream>>>(
        xb, Wt, bias, fbuf, gbuf, vt);

    // 3) flash attention partials: (4 batches x 32 qblocks x ksplit) blocks
    attn_kernel<<<NBATCH * (NTOK / QBLK) * ksplit, 256, 0, stream>>>(
        fbuf, gbuf, vt, Opart, mpart, lpart, ksplit);

    // 4) combine splits + 1/l + gamma + residual
    combine_kernel<<<(int)(((size_t)ROWS * CDIM + 255) / 256), 256, 0, stream>>>(
        Opart, mpart, lpart, x, gm, y, ksplit);
}
